// AttentionConv2D_55155970015768
// MI455X (gfx1250) — matrix operand order repaired and checked by reading, not compiled, not run on hardware
//
#include <hip/hip_runtime.h>
#include <hip/hip_bf16.h>

// ---------------------------------------------------------------------------
// AttentionConv2D on gfx1250 (MI455X): implicit-GEMM 3x3 conv via WMMA f16
// with CDNA5 async global->LDS staging (global_load_async_to_lds_b128 +
// s_wait_asynccnt), BN folded into weights at prep time, pointwise attention
// + tiny 3x3 + sigmoid as memory-bound epilogue passes.
// Shapes fixed per reference: B=4, Cin=128, H=W=256, branch channels=64.
// ---------------------------------------------------------------------------

typedef _Float16 v16h __attribute__((ext_vector_type(16)));
typedef _Float16 v8h  __attribute__((ext_vector_type(8)));
typedef float    v8f  __attribute__((ext_vector_type(8)));

#define BATCH 4
#define CIN   128
#define HH    256
#define WW    256
#define NPX   (BATCH * HH * WW)

// workspace layout (bytes)
#define WPK_HALVES  (2 * 9 * 4 * 2 * 64 * 16)   // 147456 f16 packed weights
#define BIAS_OFF    294912                       // 128 floats
#define ATTP_OFF    295424                       // 2*67 floats (folded 1x1 att)
#define ATTBUF_OFF  296192                       // 4*2*256*256 floats (2 MB)

// LDS geometry for the conv kernel (row-sliced)
#define SLAB_COLS  80     // f32 staging: cols w0-8 .. w0+71 (20 x 16B chunks)
#define CI_STRIDE  136    // f16 row buffer: 128 ci + 8 halves pad (bank stride)

__device__ __forceinline__ float grid1(int i) {
  // |i - H/2 + 0.5| / (H/2), H = 256 (even)
  return fabsf((float)i - 127.5f) * (1.0f / 128.0f);
}
__device__ __forceinline__ float polar_r(float gh, float gw) {
  float pr = sqrtf(gh * gh + gw * gw);
  const float mx = 1.4086893f;   // sqrt(2) * 127.5/128
  const float mn = 0.0055243f;   // sqrt(2) * 0.5/128
  const float k  = 2.0f / (mx - mn);
  return k * pr + (1.0f - mx * k);
}

// ---------------------------------------------------------------------------
// K0: fold BN into conv weights (-> f16, WMMA-B fragment packed layout),
//     fold conv bias + BN -> per-channel bias, fold 1x1 att conv + BN.
// packed index: ((((branch*9 + t)*4 + kb)*2 + half)*64 + co)*16 + j
//   with ci = kb*32 + half*16 + j   (matches B-fragment per-lane layout)
// ---------------------------------------------------------------------------
__global__ void prep_kernel(
    const float* __restrict__ conv_a_w, const float* __restrict__ conv_a_b,
    const float* __restrict__ conv_b_w, const float* __restrict__ conv_b_b,
    const float* __restrict__ bn_a_g, const float* __restrict__ bn_a_be,
    const float* __restrict__ bn_a_m, const float* __restrict__ bn_a_v,
    const float* __restrict__ bn_b_g, const float* __restrict__ bn_b_be,
    const float* __restrict__ bn_b_m, const float* __restrict__ bn_b_v,
    const float* __restrict__ att_a_w, const float* __restrict__ att_a_b,
    const float* __restrict__ att_b_w, const float* __restrict__ att_b_b,
    const float* __restrict__ abn_a_g, const float* __restrict__ abn_a_be,
    const float* __restrict__ abn_a_m, const float* __restrict__ abn_a_v,
    const float* __restrict__ abn_b_g, const float* __restrict__ abn_b_be,
    const float* __restrict__ abn_b_m, const float* __restrict__ abn_b_v,
    _Float16* __restrict__ wpk, float* __restrict__ biasf,
    float* __restrict__ attp)
{
  int idx = blockIdx.x * 256 + threadIdx.x;
  if (idx < WPK_HALVES) {
    int j  = idx & 15;
    int r  = idx >> 4;
    int co = r & 63;  r >>= 6;
    int half = r & 1; r >>= 1;
    int kb = r & 3;   r >>= 2;
    int t  = r % 9;
    int branch = r / 9;
    int ci = kb * 32 + half * 16 + j;
    const float* cw = branch ? conv_b_w : conv_a_w;
    const float* g  = branch ? bn_b_g  : bn_a_g;
    const float* vv = branch ? bn_b_v  : bn_a_v;
    float scale = g[co] * rsqrtf(vv[co] + 1e-3f);
    float wv = cw[(co * CIN + ci) * 9 + t];
    wpk[idx] = (_Float16)(wv * scale);
  }
  if (idx < 128) {
    int branch = idx >> 6, co = idx & 63;
    const float* g  = branch ? bn_b_g  : bn_a_g;
    const float* be = branch ? bn_b_be : bn_a_be;
    const float* mn = branch ? bn_b_m  : bn_a_m;
    const float* vv = branch ? bn_b_v  : bn_a_v;
    const float* cb = branch ? conv_b_b : conv_a_b;
    float scale = g[co] * rsqrtf(vv[co] + 1e-3f);
    biasf[idx] = (cb[co] - mn[co]) * scale + be[co];
  }
  if (idx < 132) {
    int branch = idx / 66, c = idx % 66;
    const float* aw = branch ? att_b_w : att_a_w;
    float s = (branch ? abn_b_g[0] : abn_a_g[0]) *
              rsqrtf((branch ? abn_b_v[0] : abn_a_v[0]) + 1e-5f);
    attp[branch * 67 + c] = aw[c] * s;
    if (c == 0) {
      float ab = branch ? att_b_b[0]  : att_a_b[0];
      float mn = branch ? abn_b_m[0]  : abn_a_m[0];
      float be = branch ? abn_b_be[0] : abn_a_be[0];
      attp[branch * 67 + 66] = (ab - mn) * s + be;
    }
  }
}

// ---------------------------------------------------------------------------
// K1: implicit-GEMM 3x3 conv, both branches, via v_wmma_f32_16x16x32_f16.
// Block = 256 threads (8 wave32). Tile = 64 consecutive output pixels of one
// (b,h) row. Loop over the 3 kernel rows:
//   (1) async-DMA the 128ci x 80col f32 slab of that input row into LDS
//       (global_load_async_to_lds_b128, GVS addressing, ASYNCcnt tracked)
//   (2) s_wait_asynccnt 0 + barrier
//   (3) LDS->LDS convert to f16 row buffer in WMMA-A layout (zero-pads halo)
//   (4) 3 kw taps x 4 ci-blocks x 4 N-tiles of WMMA, accumulate f32
// B fragments are contiguous 32B global reads of the prep-packed weights
// (L2-resident, ~0.3 MB total).
// ---------------------------------------------------------------------------
__global__ void __launch_bounds__(256)
conv_wmma_kernel(const float* __restrict__ x,
                 const _Float16* __restrict__ wpk,
                 const float* __restrict__ biasf,
                 float* __restrict__ out)
{
  __shared__ float    sf32[CIN * SLAB_COLS];      // 40960 B f32 staging slab
  __shared__ _Float16 sf16[66 * CI_STRIDE];       // 17952 B f16 WMMA-A row

  const int tid = threadIdx.x;
  const int b  = blockIdx.x >> 10;
  const int h  = (blockIdx.x >> 2) & 255;
  const int w0 = (blockIdx.x & 3) * 64;

  const int wave   = tid >> 5;
  const int lane   = tid & 31;
  const int mblk   = wave & 3;       // which 16-pixel sub-block
  const int branch = wave >> 2;      // 0 = a, 1 = b
  const int halfId = lane >> 4;      // K-half per WMMA layout
  const int n      = lane & 15;      // N index within a 16-wide tile

  v8f acc[4] = {};
  const _Float16* wp = wpk + (size_t)branch * 9 * 4 * 2 * 64 * 16;

  for (int kh = 0; kh < 3; ++kh) {
    const int hh = h + kh - 1;
    const bool rowOK = (unsigned)hh < 256u;

    // ---- (1) async global->LDS DMA of this input row's slab --------------
    // 128ci x 80 f32 = 2560 16B chunks; 10 chunks per thread.
    if (rowOK) {
#pragma unroll
      for (int k = 0; k < 10; ++k) {
        int id  = tid + k * 256;
        int cix = id % 20;             // 16B chunk within the 80-col slab
        int ci  = id / 20;
        int wcol = w0 - 8 + cix * 4;   // first f32 column of this chunk
        if (wcol >= 0 && wcol + 4 <= WW) {
          unsigned ldsaddr = (unsigned)(uintptr_t)(&sf32[ci * SLAB_COLS + cix * 4]);
          unsigned goff = (unsigned)(((((b * CIN + ci) * HH + hh) * WW) + wcol) * 4);
          asm volatile("global_load_async_to_lds_b128 %0, %1, %2"
                       :
                       : "v"(ldsaddr), "v"(goff), "s"(x)
                       : "memory");
        }
      }
    }
    // ---- (2) wait for DMA, sync all waves --------------------------------
    asm volatile("s_wait_asynccnt 0x0" ::: "memory");
    __syncthreads();

    // ---- (3) LDS f32 slab -> LDS f16 row buffer (zero-pad halo) ----------
    for (int idx = tid; idx < 66 * CIN; idx += 256) {
      int c  = idx % 66;               // col c <-> pixel w0 - 1 + c
      int ci = idx / 66;
      int wcol = w0 - 1 + c;
      float v = (rowOK && (unsigned)wcol < 256u)
                    ? sf32[ci * SLAB_COLS + (c + 7)]
                    : 0.0f;
      sf16[c * CI_STRIDE + ci] = (_Float16)v;
    }
    __syncthreads();

    // ---- (4) WMMA over kw taps and ci blocks -----------------------------
#pragma unroll
    for (int kw = 0; kw < 3; ++kw) {
      const int t = kh * 3 + kw;
      const int pcol = mblk * 16 + n + kw;
      const _Float16* abase = &sf16[pcol * CI_STRIDE];

      // prefetch next tap's weights (global_prefetch_b8)
      __builtin_prefetch(wp + (size_t)(((t + 1) % 9) * 4 * 2 * 64) * 16, 0, 0);

#pragma unroll
      for (int kb = 0; kb < 4; ++kb) {
        // A fragment: lane<16 holds K 0..15, lane>=16: 16..31
        v8h lo = *(const v8h*)(abase + kb * 32 + halfId * 16);
        v8h hi = *(const v8h*)(abase + kb * 32 + halfId * 16 + 8);
        v16h a = __builtin_shufflevector(lo, hi, 0, 1, 2, 3, 4, 5, 6, 7,
                                                 8, 9, 10, 11, 12, 13, 14, 15);

        const _Float16* bb =
            wp + (size_t)(((t * 4 + kb) * 2 + halfId) * 64) * 16;
#pragma unroll
        for (int nt = 0; nt < 4; ++nt) {
          v16h bf = *(const v16h*)(bb + (((nt * 16) + n) << 4));
          acc[nt] = __builtin_amdgcn_wmma_f32_16x16x32_f16(
              false, a, false, bf, (short)0, acc[nt], false, false);
        }
      }
    }
    __syncthreads();   // protect LDS buffers before next row's DMA
  }

  // ---- epilogue: add folded bias, store fa/fb into d_out (NCHW f32) ----
#pragma unroll
  for (int nt = 0; nt < 4; ++nt) {
    const int ch = branch * 64 + nt * 16 + n;
    const float bv = biasf[ch];
    float* op = out + (((size_t)b * 128 + ch) * HH + h) * WW;
#pragma unroll
    for (int r = 0; r < 8; ++r) {
      int px = w0 + mblk * 16 + halfId * 8 + r;   // M = r (+8 for upper lanes)
      op[px] = acc[nt][r] + bv;
    }
  }
}

// ---------------------------------------------------------------------------
// K2: pointwise 1x1 attention (BN folded) per pixel -> att buffer in ws.
// ---------------------------------------------------------------------------
__global__ void att_kernel(const float* __restrict__ out,
                           const float* __restrict__ attp,
                           float* __restrict__ attbuf)
{
  int idx = blockIdx.x * 256 + threadIdx.x;
  if (idx >= NPX) return;
  int b = idx >> 16;
  int h = (idx >> 8) & 255;
  int w = idx & 255;
  float gh = grid1(h), gw = grid1(w);
#pragma unroll
  for (int br = 0; br < 2; ++br) {
    const float* p = attp + br * 67;
    float dot = p[64] * gh + p[65] * gw + p[66];
    const float* f = out + (((size_t)b * 128 + br * 64) * HH + h) * WW + w;
#pragma unroll 8
    for (int c = 0; c < 64; ++c)
      dot += p[c] * f[(size_t)c * (HH * WW)];
    attbuf[(((size_t)b * 2 + br) * HH + h) * WW + w] = dot;
  }
}

// ---------------------------------------------------------------------------
// K3: 3x3 attn conv over [att, gh, gw, pr] (zero pad), sigmoid, scale,
//     multiply fa/fb in-place in d_out.
// ---------------------------------------------------------------------------
__global__ void attmap_kernel(float* __restrict__ out,
                              const float* __restrict__ attbuf,
                              const float* __restrict__ attn_a_w,
                              const float* __restrict__ attn_b_w,
                              const float* __restrict__ scale1,
                              const float* __restrict__ scale2)
{
  int idx = blockIdx.x * 256 + threadIdx.x;
  if (idx >= NPX) return;
  int b = idx >> 16;
  int h = (idx >> 8) & 255;
  int w = idx & 255;

#pragma unroll
  for (int br = 0; br < 2; ++br) {
    const float* wv = br ? attn_b_w : attn_a_w;   // [4][3][3]
    float s = 0.0f;
#pragma unroll
    for (int kh = 0; kh < 3; ++kh) {
#pragma unroll
      for (int kw = 0; kw < 3; ++kw) {
        int qh = h + kh - 1, qw = w + kw - 1;
        if ((unsigned)qh < 256u && (unsigned)qw < 256u) {
          float att = attbuf[(((size_t)b * 2 + br) * HH + qh) * WW + qw];
          float gh = grid1(qh), gw = grid1(qw);
          float pr = polar_r(gh, gw);
          int o = kh * 3 + kw;
          s += att * wv[o] + gh * wv[9 + o] + gw * wv[18 + o] + pr * wv[27 + o];
        }
      }
    }
    float map = 1.0f / (1.0f + __expf(-s));
    float m = map * (br ? scale2[0] : scale1[0]);
    float* f = out + (((size_t)b * 128 + br * 64) * HH + h) * WW + w;
#pragma unroll 8
    for (int c = 0; c < 64; ++c)
      f[(size_t)c * (HH * WW)] *= m;
  }
}

// ---------------------------------------------------------------------------
extern "C" void kernel_launch(void* const* d_in, const int* in_sizes, int n_in,
                              void* d_out, int out_size, void* d_ws,
                              size_t ws_size, hipStream_t stream) {
  (void)in_sizes; (void)n_in; (void)out_size; (void)ws_size;

  const float* x        = (const float*)d_in[0];
  const float* conv_a_w = (const float*)d_in[1];
  const float* conv_a_b = (const float*)d_in[2];
  const float* conv_b_w = (const float*)d_in[3];
  const float* conv_b_b = (const float*)d_in[4];
  const float* bn_a_g   = (const float*)d_in[5];
  const float* bn_a_be  = (const float*)d_in[6];
  const float* bn_a_m   = (const float*)d_in[7];
  const float* bn_a_v   = (const float*)d_in[8];
  const float* bn_b_g   = (const float*)d_in[9];
  const float* bn_b_be  = (const float*)d_in[10];
  const float* bn_b_m   = (const float*)d_in[11];
  const float* bn_b_v   = (const float*)d_in[12];
  const float* att_a_w  = (const float*)d_in[13];
  const float* att_a_b  = (const float*)d_in[14];
  const float* att_b_w  = (const float*)d_in[15];
  const float* att_b_b  = (const float*)d_in[16];
  const float* abn_a_g  = (const float*)d_in[17];
  const float* abn_a_be = (const float*)d_in[18];
  const float* abn_a_m  = (const float*)d_in[19];
  const float* abn_a_v  = (const float*)d_in[20];
  const float* abn_b_g  = (const float*)d_in[21];
  const float* abn_b_be = (const float*)d_in[22];
  const float* abn_b_m  = (const float*)d_in[23];
  const float* abn_b_v  = (const float*)d_in[24];
  const float* attn_a_w = (const float*)d_in[25];
  const float* attn_b_w = (const float*)d_in[26];
  const float* scale1   = (const float*)d_in[27];
  const float* scale2   = (const float*)d_in[28];

  _Float16* wpk   = (_Float16*)d_ws;
  float*    biasf = (float*)((char*)d_ws + BIAS_OFF);
  float*    attp  = (float*)((char*)d_ws + ATTP_OFF);
  float*    attbuf= (float*)((char*)d_ws + ATTBUF_OFF);
  float*    out   = (float*)d_out;

  // K0: fold BN, pack weights into WMMA-B fragment layout
  prep_kernel<<<(WPK_HALVES + 255) / 256, 256, 0, stream>>>(
      conv_a_w, conv_a_b, conv_b_w, conv_b_b,
      bn_a_g, bn_a_be, bn_a_m, bn_a_v,
      bn_b_g, bn_b_be, bn_b_m, bn_b_v,
      att_a_w, att_a_b, att_b_w, att_b_b,
      abn_a_g, abn_a_be, abn_a_m, abn_a_v,
      abn_b_g, abn_b_be, abn_b_m, abn_b_v,
      wpk, biasf, attp);

  // K1: WMMA implicit-GEMM conv (both branches) -> fa/fb in d_out
  conv_wmma_kernel<<<BATCH * HH * (WW / 64), 256, 0, stream>>>(
      x, wpk, biasf, out);

  // K2: folded 1x1 attention per pixel
  att_kernel<<<NPX / 256, 256, 0, stream>>>(out, attp, attbuf);

  // K3: 3x3 attn conv + sigmoid + in-place scaling of d_out
  attmap_kernel<<<NPX / 256, 256, 0, stream>>>(
      out, attbuf, attn_a_w, attn_b_w, scale1, scale2);
}
